// Alignment_2396591751216
// MI455X (gfx1250) — matrix operand order repaired and checked by reading, not compiled, not run on hardware
//
#include <hip/hip_runtime.h>
#include <hip/hip_bf16.h>

// ---- problem constants (match reference) ----
#define B_SZ   16
#define T_IN   512
#define ENC2   512               // x channels
#define SPK_D  64
#define KDIM   (ENC2 + SPK_D)    // 576
#define ND     512               // ALIGN_D
#define MAX_Y  4096
#define OUTC   (ND + 64 + 64 + 1)  // 641

// ---- GEMM tiling ----
#define TM   128                 // block M tile
#define TN   128                 // block N tile
#define TK   32                  // K step (bf16 WMMA K)
#define NJ   (TN / 16)           // N sub-tiles per wave = 8
#define LDA  34                  // padded LDS stride in bf16 units (17 banks, coprime 64)

typedef __attribute__((ext_vector_type(16))) __bf16       v16bf;
typedef __attribute__((ext_vector_type(8)))  float        v8f;
typedef __attribute__((ext_vector_type(8)))  unsigned int v8u;

// -----------------------------------------------------------------------------
// Kernel 1: h = concat(x, emb[speaker]) @ W_lin + b_lin
// M=8192, N=512, K=576. 256-thread block computes a 128x128 tile:
// 8 waves, each wave = one 16-row M-tile x 8 N-tiles (8 WMMAs per K-step,
// A fragment reused across all 8). A and B staged through LDS as bf16 with
// coalesced global loads; fragment reads are aligned b32 LDS loads.
// Global traffic: ~150 MB total (vs ~1.2 GB untiled), fully L2-resident.
// -----------------------------------------------------------------------------
__global__ __launch_bounds__(256)
void align_gemm_wmma(const float* __restrict__ x,
                     const int*   __restrict__ speaker,
                     const float* __restrict__ emb,
                     const float* __restrict__ W,      // [KDIM, ND] row-major
                     const float* __restrict__ bias,   // [ND]
                     float*       __restrict__ h)      // [M, ND]
{
    __shared__ unsigned short As[TM * LDA];  // [row][k]   bf16
    __shared__ unsigned short Bs[TN * LDA];  // [n][k]     bf16 (transposed)

    const int tid  = threadIdx.x;
    const int lane = tid & 31;
    const int wave = tid >> 5;     // 0..7 -> M sub-tile within block
    const int l15  = lane & 15;
    const int lhi  = lane >> 4;    // 0 for lanes 0-15, 1 for lanes 16-31

    const int m0 = blockIdx.y * TM;
    const int n0 = blockIdx.x * TN;

    v8f acc[NJ] = {};

    for (int kk = 0; kk < KDIM; kk += TK) {
        // ---- stage A (TM x TK) as bf16; fuse concat(x, emb gather) ----
        // flat = row*TK + k ; 32 consecutive lanes cover one 128 B row: coalesced
#pragma unroll
        for (int i = 0; i < (TM * TK) / 256; ++i) {
            const int f  = tid + i * 256;
            const int r  = f >> 5;          // 0..127
            const int k  = f & (TK - 1);    // 0..31
            const int gk = kk + k;
            const int gr = m0 + r;
            const float a = (gk < ENC2)
                ? x[(size_t)gr * ENC2 + gk]
                : emb[(size_t)speaker[gr] * SPK_D + (gk - ENC2)];
            const __bf16 ab = (__bf16)a;
            As[r * LDA + k] = __builtin_bit_cast(unsigned short, ab);
        }
        // ---- stage B (TK x TN) transposed to [n][k] ----
        // flat = k*TN + n ; consecutive lanes -> consecutive n: coalesced
#pragma unroll
        for (int i = 0; i < (TK * TN) / 256; ++i) {
            const int f = tid + i * 256;
            const int k = f >> 7;           // 0..31
            const int n = f & (TN - 1);     // 0..127
            const float wv = W[(size_t)(kk + k) * ND + n0 + n];
            const __bf16 wb = (__bf16)wv;
            Bs[n * LDA + k] = __builtin_bit_cast(unsigned short, wb);
        }
        // prefetch next K-slab of W (global_prefetch_b8 path)
        if (kk + TK < KDIM)
            __builtin_prefetch(&W[(size_t)(kk + TK + (tid >> 7)) * ND + n0 + (tid & 127)], 0, 1);
        __syncthreads();

        // ---- A fragment (16x32 bf16, ISA 7.12.2 layout), reused by 8 WMMAs ----
        v8u au;
#pragma unroll
        for (int v = 0; v < 8; ++v) {
            const int kA = ((v & 3) << 1) + ((v >> 2) << 4) + (lhi << 3); // even
            au[v] = *(const unsigned int*)&As[(wave * 16 + l15) * LDA + kA];
        }
        const v16bf af = __builtin_bit_cast(v16bf, au);

        // ---- 8 N sub-tiles: B fragment from LDS, one WMMA each ----
#pragma unroll
        for (int j = 0; j < NJ; ++j) {
            v8u bu;
#pragma unroll
            for (int v = 0; v < 8; ++v) {
                const int kB = ((v & 3) << 1) + ((v >> 2) << 4) + (lhi << 3); // even
                bu[v] = *(const unsigned int*)&Bs[(j * 16 + l15) * LDA + kB];
            }
            const v16bf bf = __builtin_bit_cast(v16bf, bu);
            acc[j] = __builtin_amdgcn_wmma_f32_16x16x32_bf16(
                         false, af, false, bf, (short)0, acc[j], false, false);
        }
        __syncthreads();
    }

    // ---- epilogue: bias add + store (lanes 0-15 -> consecutive cols) ----
#pragma unroll
    for (int j = 0; j < NJ; ++j) {
        const int col = n0 + j * 16 + l15;
        const float bv = bias[col];
#pragma unroll
        for (int v = 0; v < 8; ++v) {
            const int row = m0 + wave * 16 + v + (lhi << 3);
            h[(size_t)row * ND + col] = acc[j][v] + bv;
        }
    }
}

// -----------------------------------------------------------------------------
// Kernel 2: per-batch inclusive scan of durations
// -----------------------------------------------------------------------------
__global__ __launch_bounds__(512)
void align_scan(const int* __restrict__ dur, int* __restrict__ cum)
{
    __shared__ int s[T_IN];
    const int b = blockIdx.x;
    const int t = threadIdx.x;
    s[t] = dur[b * T_IN + t];
    __syncthreads();
#pragma unroll
    for (int off = 1; off < T_IN; off <<= 1) {
        const int v = (t >= off) ? s[t - off] : 0;
        __syncthreads();
        s[t] += v;
        __syncthreads();
    }
    cum[b * T_IN + t] = s[t];
}

// -----------------------------------------------------------------------------
// Kernel 3: fused length-regulate + f0/rmse rank-1 projections + pos concat.
// One 256-thread block per (b, y) output row; 641 coalesced channel stores.
// HBM-store-bound: 168 MB out; h-gather reads hit L2 (16.8 MB << 192 MB).
// -----------------------------------------------------------------------------
__global__ __launch_bounds__(256)
void align_expand(const float* __restrict__ h,
                  const int*   __restrict__ cum,
                  const float* __restrict__ f0,
                  const float* __restrict__ rmse,
                  const float* __restrict__ position,
                  const float* __restrict__ W_f0,   const float* __restrict__ b_f0,
                  const float* __restrict__ W_rmse, const float* __restrict__ b_rmse,
                  float* __restrict__ out)
{
    const int y = blockIdx.x;
    const int b = blockIdx.y;
    const int* __restrict__ c = cum + b * T_IN;

    const int  total = c[T_IN - 1];
    const bool valid = (y < total);

    // searchsorted(c, y, side='right'): first i with c[i] > y
    int lo = 0, hi = T_IN;
    while (lo < hi) {
        const int mid = (lo + hi) >> 1;
        if (c[mid] <= y) lo = mid + 1; else hi = mid;
    }
    const int idx = (lo < T_IN) ? lo : (T_IN - 1);

    const float* __restrict__ hrow = h + ((size_t)b * T_IN + idx) * ND;
    const size_t obase = ((size_t)b * MAX_Y + y) * (size_t)OUTC;

    const float f0v = f0[b * MAX_Y + y];
    const float rv  = rmse[b * MAX_Y + y];
    const float pv  = position[b * MAX_Y + y];

    for (int ch = threadIdx.x; ch < OUTC; ch += 256) {
        float val;
        if (ch < ND)            val = valid ? hrow[ch] : 0.0f;
        else if (ch < ND + 64)  val = fmaf(f0v, W_f0[ch - ND], b_f0[ch - ND]);
        else if (ch < ND + 128) val = fmaf(rv, W_rmse[ch - (ND + 64)], b_rmse[ch - (ND + 64)]);
        else                    val = pv;
        out[obase + ch] = val;
    }
}

// -----------------------------------------------------------------------------
extern "C" void kernel_launch(void* const* d_in, const int* in_sizes, int n_in,
                              void* d_out, int out_size, void* d_ws, size_t ws_size,
                              hipStream_t stream)
{
    (void)in_sizes; (void)n_in; (void)out_size; (void)ws_size;

    const float* x        = (const float*)d_in[0];   // [16,512,512]
    const int*   speaker  = (const int*)  d_in[1];   // [16,512]
    const int*   duration = (const int*)  d_in[2];   // [16,512]
    const float* f0       = (const float*)d_in[3];   // [16,4096]
    const float* rmse     = (const float*)d_in[4];   // [16,4096]
    const float* position = (const float*)d_in[5];   // [16,4096]
    /* d_in[6] = max_y_len (compile-time constant MAX_Y) */
    const float* emb      = (const float*)d_in[7];   // [128,64]
    const float* W_lin    = (const float*)d_in[8];   // [576,512]
    const float* b_lin    = (const float*)d_in[9];   // [512]
    const float* W_f0     = (const float*)d_in[10];  // [1,64]
    const float* b_f0     = (const float*)d_in[11];  // [64]
    const float* W_rmse   = (const float*)d_in[12];  // [1,64]
    const float* b_rmse   = (const float*)d_in[13];  // [64]

    float* out = (float*)d_out;                      // [16,4096,641]

    // workspace: [0, 32KB) cumsum ints; [64KB, 64KB+16.8MB) h fp32
    int*   cum = (int*)d_ws;
    float* h   = (float*)((char*)d_ws + (64u << 10));

    // 1) GEMM: grid (N/TN, M/TM) = (4, 64), 256 threads/block
    dim3 ggrid(ND / TN, (B_SZ * T_IN) / TM);
    align_gemm_wmma<<<ggrid, 256, 0, stream>>>(x, speaker, emb, W_lin, b_lin, h);

    // 2) duration cumsum per batch row
    align_scan<<<B_SZ, T_IN, 0, stream>>>(duration, cum);

    // 3) fused expand + concat
    dim3 egrid(MAX_Y, B_SZ);
    align_expand<<<egrid, 256, 0, stream>>>(h, cum, f0, rmse, position,
                                            W_f0, b_f0, W_rmse, b_rmse, out);
}
